// DecoderSelfAttention_17575006175501
// MI455X (gfx1250) — compile-verified, matrix operand order checked by reading, not run on hardware
//
#include <hip/hip_runtime.h>

// ---------------------------------------------------------------------------
// Single-head causal self-attention, B=8, S=2048, D=1024 (fp32 in/out).
// Strategy: bf16 WMMA (v_wmma_f32_16x16x32_bf16) for all GEMMs, fp32 softmax.
//   Kernel 1: QKV projection  x[16384,1024] @ W[1024,1024] -> bf16 Q,K, V^T
//             (double-buffered LDS, fragment-ready layout, packed staging)
//   Kernel 2: per-(batch, 16-row q tile) attention with full score row in LDS
// ---------------------------------------------------------------------------

typedef __attribute__((ext_vector_type(16))) __bf16 v16bf;
typedef __attribute__((ext_vector_type(8)))  __bf16 v8bf;
typedef __attribute__((ext_vector_type(4)))  __bf16 v4bf;
typedef __attribute__((ext_vector_type(8)))  float  v8f;
typedef __attribute__((ext_vector_type(4)))  float  v4f;

#define B_   8
#define S_   2048
#define D_   1024
#define SROW 2052   // padded fp32 score-row stride: 2052*4B = 513*16B

static __device__ __forceinline__ v8f zero8() {
  v8f z = {0.f, 0.f, 0.f, 0.f, 0.f, 0.f, 0.f, 0.f};
  return z;
}

static __device__ __forceinline__ void wmma_bf16(v16bf a, v16bf b, v8f& c) {
  c = __builtin_amdgcn_wmma_f32_16x16x32_bf16(false, a, false, b, (short)0, c,
                                              false, false);
}

// Load a 16x32 (16-bit) WMMA A/B fragment directly from a row-major bf16
// matrix.  rowBase points at element (row0, k0).  Per-lane: row = lane&15;
// low half-wave takes k {0..7,16..23}, high half k {8..15,24..31}
// -> two contiguous 16-byte loads per lane.
static __device__ __forceinline__ v16bf frag_ld(const __bf16* rowBase,
                                                int rowStride, int lane) {
  const int r = lane & 15;
  const int half = lane >> 4;
  const __bf16* p = rowBase + (size_t)r * (size_t)rowStride + (half << 3);
  v8bf lo = *(const v8bf*)p;
  v8bf hi = *(const v8bf*)(p + 16);
  v16bf f;
#pragma unroll
  for (int i = 0; i < 8; ++i) { f[i] = lo[i]; f[i + 8] = hi[i]; }
  return f;
}

// ---------------------------------------------------------------------------
// Kernel 1: QKV projection.  blockIdx.z selects {Wq,Wk,Wv}.
// 128x128 output tile per 256-thread block; K panels of 32 staged through
// double-buffered LDS in fragment-ready layout so global loads of panel i+1
// overlap the WMMA chain of panel i.
// ---------------------------------------------------------------------------

// Load one 128x32 x-panel chunk set + one 32x128 W-panel block into registers.
static __device__ __forceinline__ void panel_load(
    v4f va[4], v4f vb[4], const float* __restrict__ x,
    const float* __restrict__ W, int kb, int tid, int mbase, int nbase) {
#pragma unroll
  for (int j = 0; j < 4; ++j) {           // A: 4 float4 chunks / thread
    int c = tid + j * 256;                // 1024 chunks total
    int mm = c >> 3;                      // 0..127
    int kc = (c & 7) << 2;                // 0..28
    va[j] = *(const v4f*)(x + (size_t)(mbase + mm) * D_ + kb + kc);
  }
  int kk0 = (tid >> 5) << 2;              // B: thread owns 4k x 4n block
  int nc0 = (tid & 31) << 2;
#pragma unroll
  for (int i = 0; i < 4; ++i)
    vb[i] = *(const v4f*)(W + (size_t)(kb + kk0 + i) * D_ + nbase + nc0);
}

// Convert + scatter registers into fragment-ready LDS (packed b64 stores).
static __device__ __forceinline__ void panel_store(
    __bf16* aBuf, __bf16* bBuf, const v4f va[4], const v4f vb[4], int tid) {
#pragma unroll
  for (int j = 0; j < 4; ++j) {
    int c = tid + j * 256;
    int mm = c >> 3;
    int kc = (c & 7) << 2;
    int half = (kc >> 3) & 1;
    int idx = (kc & 7) + ((kc & 16) >> 1);
    __bf16* d = aBuf + ((mm >> 4) << 9) + (((mm & 15) + (half << 4)) << 4) + idx;
    v4bf pk;
    pk[0] = (__bf16)va[j].x; pk[1] = (__bf16)va[j].y;
    pk[2] = (__bf16)va[j].z; pk[3] = (__bf16)va[j].w;
    *(v4bf*)d = pk;                       // ds_store_b64
  }
  int kk0 = (tid >> 5) << 2;
  int nc0 = (tid & 31) << 2;
  int halfB = (kk0 >> 3) & 1;
  int idxB = (kk0 & 7) + ((kk0 & 16) >> 1);  // k0..k0+3 -> idxB..idxB+3
#pragma unroll
  for (int i = 0; i < 4; ++i) {
    int n = nc0 + i;
    __bf16* d = bBuf + ((n >> 4) << 9) + (((n & 15) + (halfB << 4)) << 4) + idxB;
    v4bf pk;
    pk[0] = (__bf16)vb[0][i]; pk[1] = (__bf16)vb[1][i];
    pk[2] = (__bf16)vb[2][i]; pk[3] = (__bf16)vb[3][i];
    *(v4bf*)d = pk;                       // ds_store_b64
  }
}

__global__ __launch_bounds__(256) void qkv_proj_kernel(
    const float* __restrict__ x, const float* __restrict__ Wq,
    const float* __restrict__ Wk, const float* __restrict__ Wv,
    __bf16* __restrict__ Qb, __bf16* __restrict__ Kb, __bf16* __restrict__ Vt) {
  __shared__ __bf16 aBuf[2][8 * 512];  // double-buffered, fragment-ready
  __shared__ __bf16 bBuf[2][8 * 512];

  const int tid = threadIdx.x;
  const int lane = tid & 31;
  const int wave = tid >> 5;
  const int mbase = blockIdx.x * 128;
  const int nbase = blockIdx.y * 128;
  const int z = blockIdx.z;
  const float* W = (z == 0) ? Wq : (z == 1) ? Wk : Wv;

  v8f acc[8];
#pragma unroll
  for (int t = 0; t < 8; ++t) acc[t] = zero8();

  v4f va[4], vb[4];
  panel_load(va, vb, x, W, 0, tid, mbase, nbase);

  int cur = 0;
  for (int kb = 0; kb < D_; kb += 32) {
    panel_store(aBuf[cur], bBuf[cur], va, vb, tid);
    __syncthreads();
    if (kb + 32 < D_)                       // overlap next panel's HBM loads
      panel_load(va, vb, x, W, kb + 32, tid, mbase, nbase);
    v16bf aF = *(const v16bf*)(aBuf[cur] + (wave << 9) + (lane << 4));
#pragma unroll
    for (int t = 0; t < 8; ++t) {
      v16bf bF = *(const v16bf*)(bBuf[cur] + (t << 9) + (lane << 4));
      wmma_bf16(aF, bF, acc[t]);
    }
    cur ^= 1;
  }

  // ---- epilogue: C layout -> vgpr r holds row r (lanes 0-15) / 8+r (16-31)
  const int half = lane >> 4;
  const int nl = lane & 15;
#pragma unroll
  for (int t = 0; t < 8; ++t) {
#pragma unroll
    for (int r = 0; r < 8; ++r) {
      int row = mbase + (wave << 4) + (half << 3) + r;  // flat (b*S + s)
      int col = nbase + (t << 4) + nl;
      __bf16 val = (__bf16)acc[t][r];
      if (z == 0) {
        Qb[(size_t)row * D_ + col] = val;
      } else if (z == 1) {
        Kb[(size_t)row * D_ + col] = val;
      } else {
        // V stored transposed: Vt[b][dv][s]  (makes P*V B-frags contiguous)
        int b = row >> 11, s = row & 2047;
        Vt[((size_t)b << 21) + ((size_t)col << 11) + (size_t)s] = val;
      }
    }
  }
}

// ---------------------------------------------------------------------------
// Kernel 2: attention.  One block = (batch b, 16 query rows).  Full causal
// score row kept in LDS fp32 (dynamic shared ~128KB; 2 WGs fit in 320KB WGP).
// ---------------------------------------------------------------------------
__global__ __launch_bounds__(256) void attn_kernel(
    const __bf16* __restrict__ Qb, const __bf16* __restrict__ Kb,
    const __bf16* __restrict__ Vt, float* __restrict__ out) {
  extern __shared__ float smem[];
  float* sSum = smem;        // 16 row sums
  float* sBuf = smem + 16;   // 16 x SROW scores

  const int tid = threadIdx.x;
  const int lane = tid & 31;
  const int wave = tid >> 5;
  const int half = lane >> 4;
  const int nl = lane & 15;

  const int q0 = blockIdx.x << 4;
  const int b = blockIdx.y;
  const int kvTiles = (q0 >> 4) + 1;             // causal: tiles 0..q0/16
  const int padTiles = (kvTiles + 1) & ~1;       // round up to K=32 chunks
  const int padEnd = padTiles << 4;

  const __bf16* Qrow = Qb + (size_t)(b * S_ + q0) * D_;
  const __bf16* Kbase = Kb + (size_t)b * S_ * D_;
  const __bf16* Vbase = Vt + ((size_t)b << 21);

  // ---------------- phase 1: scores S = Q K^T / sqrt(D) + causal mask -----
  for (int tile = wave; tile < padTiles; tile += 8) {   // wave-uniform
    const int kvb = tile << 4;
    if (tile < kvTiles) {
      const __bf16* Krow = Kbase + (size_t)kvb * D_;
      if (tile + 8 < kvTiles)  // prefetch this wave's next K tile
        __builtin_prefetch(Krow + (size_t)(8 * 16) * D_ + (size_t)nl * D_, 0, 3);
      v8f acc = zero8();
#pragma unroll 4
      for (int d = 0; d < D_; d += 32) {
        v16bf aF = frag_ld(Qrow + d, D_, lane);          // Q 16x32
        v16bf bF = frag_ld(Krow + d, D_, lane);          // K^T 32x16
        wmma_bf16(aF, bF, acc);
      }
      const int kv = kvb + nl;
#pragma unroll
      for (int r = 0; r < 8; ++r) {
        int row = (half << 3) + r;
        int q = q0 + row;
        float v = acc[r] * 0.03125f;   // 1/sqrt(1024)
        sBuf[row * SROW + kvb + nl] = (kv <= q) ? v : -3.0e38f;
      }
    } else {
#pragma unroll
      for (int r = 0; r < 8; ++r)
        sBuf[((half << 3) + r) * SROW + kvb + nl] = -3.0e38f;
    }
  }
  __syncthreads();

  // ---------------- phase 2: softmax (exp kept unnormalized in LDS) -------
  // Vectorized: 4 floats/lane/trip; padEnd % 32 == 0, chunks 16B-aligned.
  for (int rr = wave * 2; rr < wave * 2 + 2; ++rr) {
    float* row = sBuf + rr * SROW;
    v4f m4 = {-3.0e38f, -3.0e38f, -3.0e38f, -3.0e38f};
    for (int c = lane << 2; c < padEnd; c += 128) {
      v4f v = *(const v4f*)(row + c);
      m4.x = fmaxf(m4.x, v.x); m4.y = fmaxf(m4.y, v.y);
      m4.z = fmaxf(m4.z, v.z); m4.w = fmaxf(m4.w, v.w);
    }
    float mx = fmaxf(fmaxf(m4.x, m4.y), fmaxf(m4.z, m4.w));
#pragma unroll
    for (int off = 16; off; off >>= 1) mx = fmaxf(mx, __shfl_xor(mx, off, 32));
    v4f s4 = {0.f, 0.f, 0.f, 0.f};
    for (int c = lane << 2; c < padEnd; c += 128) {
      v4f v = *(const v4f*)(row + c);
      v4f e;
      e.x = __expf(v.x - mx); e.y = __expf(v.y - mx);
      e.z = __expf(v.z - mx); e.w = __expf(v.w - mx);
      *(v4f*)(row + c) = e;
      s4.x += e.x; s4.y += e.y; s4.z += e.z; s4.w += e.w;
    }
    float sum = (s4.x + s4.y) + (s4.z + s4.w);
#pragma unroll
    for (int off = 16; off; off >>= 1) sum += __shfl_xor(sum, off, 32);
    if (lane == 0) sSum[rr] = sum;
  }
  __syncthreads();

  // ---------------- phase 3: O = P V   (wave w owns 128 V columns) --------
  v8f o[8];
#pragma unroll
  for (int t = 0; t < 8; ++t) o[t] = zero8();
  const int dvb0 = wave << 7;

  for (int kc = 0; kc < padEnd; kc += 32) {
    // A-frag: exp scores, fp32 LDS -> bf16.  pr is 16B-aligned
    // (SROW*4B = 513*16B), two contiguous 8-float runs per lane.
    const float* pr = sBuf + nl * SROW + kc + (half << 3);
    v4f a0 = *(const v4f*)(pr);
    v4f a1 = *(const v4f*)(pr + 4);
    v4f b0 = *(const v4f*)(pr + 16);
    v4f b1 = *(const v4f*)(pr + 20);
    v16bf aF;
#pragma unroll
    for (int i = 0; i < 4; ++i) {
      aF[i]      = (__bf16)a0[i];
      aF[i + 4]  = (__bf16)a1[i];
      aF[i + 8]  = (__bf16)b0[i];
      aF[i + 12] = (__bf16)b1[i];
    }
    // Preload all 8 V fragments, then back-to-back WMMA.
    v16bf bF[8];
#pragma unroll
    for (int t = 0; t < 8; ++t)
      bF[t] = frag_ld(Vbase + (size_t)(dvb0 + (t << 4)) * (size_t)S_ + kc,
                      S_, lane);
#pragma unroll
    for (int t = 0; t < 8; ++t) wmma_bf16(aF, bF[t], o[t]);
  }

  float inv[8];
#pragma unroll
  for (int r = 0; r < 8; ++r) inv[r] = 1.0f / sSum[(half << 3) + r];
#pragma unroll
  for (int t = 0; t < 8; ++t) {
#pragma unroll
    for (int r = 0; r < 8; ++r) {
      int row = q0 + (half << 3) + r;
      int col = dvb0 + (t << 4) + nl;
      out[(size_t)(b * S_ + row) * D_ + col] = o[t][r] * inv[r];
    }
  }
}

// ---------------------------------------------------------------------------
extern "C" void kernel_launch(void* const* d_in, const int* in_sizes, int n_in,
                              void* d_out, int out_size, void* d_ws,
                              size_t ws_size, hipStream_t stream) {
  const float* x  = (const float*)d_in[0];
  const float* Wq = (const float*)d_in[1];
  const float* Wk = (const float*)d_in[2];
  const float* Wv = (const float*)d_in[3];

  // Workspace: bf16 Q, K (row-major) and V^T ([b][dv][s]) — 3 x 32 MB.
  __bf16* Qb = (__bf16*)d_ws;
  __bf16* Kb = Qb + (size_t)B_ * S_ * D_;
  __bf16* Vt = Kb + (size_t)B_ * S_ * D_;
  float* out = (float*)d_out;

  dim3 gp(16384 / 128, D_ / 128, 3);  // (M tiles, N tiles, {Q,K,V})
  qkv_proj_kernel<<<gp, 256, 0, stream>>>(x, Wq, Wk, Wv, Qb, Kb, Vt);

  size_t smem = (size_t)(16 + 16 * SROW) * sizeof(float);  // ~128 KB
  dim3 ga(S_ / 16, B_);
  attn_kernel<<<ga, 256, smem, stream>>>(Qb, Kb, Vt, out);
}
